// SelfAttention_43533788512748
// MI455X (gfx1250) — matrix operand order repaired and checked by reading, not compiled, not run on hardware
//
#include <hip/hip_runtime.h>
#include <hip/hip_bf16.h>

// ---------------------------------------------------------------------------
// Problem constants (from reference): inputs (L,B,H), L=256, B=8, H=512.
// x[b,l,h] = inputs[l,b,h];  m := b*256 + l  (M dimension of GEMM, 2048 rows)
// p1[m,o] = sum_k x[m,k] * attn_W[o, k]        (o,k in [0,512))
// p2[m,o] = sum_k x[m,k] * attn_W[o, 512+k]
// e[b,i,j] = sum_h tanh(p1[b*256+i,h] + p2[b*256+j,h] + attn_b[h]) * sW[h] + sb
// out[b,i,j] = softmax_i(e[b,i,j])
// ---------------------------------------------------------------------------

typedef __attribute__((ext_vector_type(16))) _Float16 v16h;
typedef __attribute__((ext_vector_type(8)))  float    v8f;

#define LQ 256
#define BQ 8
#define HQ 512
#define MROWS 2048   // B*L

// ============================= Kernel A ====================================
// WMMA projection: P[0][m][o] = p1, P[1][m][o] = p2.
// One wave (32 lanes) per 16x16 output tile; K=512 -> 16 x v_wmma_f32_16x16x32_f16.
// Layouts per CDNA5 ISA 7.12.2:
//   A 16x32 f16 : lane holds row M=lane&15; VGPR v -> K = 2v + 8*(v>=4) + 8*(lane>>4)
//   B 32x16 f16 : lane holds col N=lane&15; VGPR v -> K = 2v + 8*(v>=4) + 8*(lane>>4)
//   D 16x16 f32 : lane holds col N=lane&15; VGPR v -> row M = v + 8*(lane>>4)
__global__ __launch_bounds__(256) void proj_wmma(const float* __restrict__ inputs,
                                                 const float* __restrict__ attn_W,
                                                 float* __restrict__ P) {
  const int lane = threadIdx.x & 31;
  const int wave = threadIdx.x >> 5;
  const int tile = blockIdx.x * 8 + wave;     // 0..8191
  const int half = tile >> 12;                // 0 -> p1, 1 -> p2
  const int rem  = tile & 4095;
  const int m0   = (rem >> 5) << 4;           // 128 M-tiles
  const int o0   = (rem & 31) << 4;           // 32  O-tiles

  const int n  = lane & 15;
  const int hi = lane >> 4;

  // A source row: X[m0+n, :] = inputs[(l*B + b)*H + :]
  const int mrow = m0 + n;
  const int xb   = mrow >> 8;                 // / 256
  const int xl   = mrow & 255;
  const float* Arow = inputs + (size_t)(xl * BQ + xb) * HQ;

  // B source row: attn_W[o0+n, half*512 + :]   (row-major (512,1024))
  const float* Brow = attn_W + (size_t)(o0 + n) * (2 * HQ) + half * HQ;

  v8f acc = {};
  #pragma unroll
  for (int kk = 0; kk < 16; ++kk) {
    const int k0 = kk * 32;
    v16h a, bm;
    #pragma unroll
    for (int v = 0; v < 8; ++v) {
      const int ka = 2 * v + ((v >= 4) ? 8 : 0) + 8 * hi;   // A K-index (even)
      const float2 fa = *(const float2*)(Arow + k0 + ka);
      a[2 * v]     = (_Float16)fa.x;
      a[2 * v + 1] = (_Float16)fa.y;
      const int kb = 2 * v + ((v >= 4) ? 8 : 0) + 8 * hi;   // B K-index (even)
      const float2 fb = *(const float2*)(Brow + k0 + kb);
      bm[2 * v]     = (_Float16)fb.x;
      bm[2 * v + 1] = (_Float16)fb.y;
    }
    // 8 args: (neg_a, A, neg_b, B, c_mod, C, reuse_a, reuse_b)
    acc = __builtin_amdgcn_wmma_f32_16x16x32_f16(false, a, false, bm,
                                                 (short)0, acc, false, false);
  }

  float* Pd = P + (size_t)half * ((size_t)MROWS * HQ);
  #pragma unroll
  for (int v = 0; v < 8; ++v) {
    Pd[(size_t)(m0 + v + 8 * hi) * HQ + (o0 + n)] = acc[v];
  }
}

// ============================= Kernel B ====================================
__device__ __forceinline__ float fast_tanhf(float x) {
#if __has_builtin(__builtin_amdgcn_tanhf)
  return __builtin_amdgcn_tanhf(x);          // v_tanh_f32 on gfx1250
#else
  const float e2 = __expf(2.0f * x);         // v_exp_f32 path
  return (e2 - 1.0f) / (e2 + 1.0f);
#endif
}

// One block per (b, j): thread t owns i=t. Fused tanh-dot + softmax over i.
__global__ __launch_bounds__(256) void energy_softmax(
    const float* __restrict__ P,        // [2][2048][512] (p1, p2)
    const float* __restrict__ attn_b,   // [512]
    const float* __restrict__ score_W,  // [512]
    const float* __restrict__ score_b,  // [1]
    float* __restrict__ out) {          // [8][256][256]
  __shared__ __align__(16) float q[HQ];    // p2[b,j,:] + attn_b
  __shared__ __align__(16) float sw[HQ];   // score_W
  __shared__ float red[256];

  const int t = threadIdx.x;
  const int j = blockIdx.x & 255;
  const int b = blockIdx.x >> 8;

  const float* P1 = P;
  const float* P2 = P + (size_t)MROWS * HQ;
  const float* p2row = P2 + (size_t)(b * 256 + j) * HQ;

  for (int h = t; h < HQ; h += 256) {
    q[h]  = p2row[h] + attn_b[h];
    sw[h] = score_W[h];
  }
  __syncthreads();

  const float* p1row = P1 + (size_t)(b * 256 + t) * HQ;
  float e = score_b[0];
  #pragma unroll 4
  for (int h = 0; h < HQ; h += 4) {
    const float4 x4 = *(const float4*)(p1row + h);
    const float4 q4 = *(const float4*)(q + h);
    const float4 s4 = *(const float4*)(sw + h);
    e += s4.x * fast_tanhf(x4.x + q4.x);
    e += s4.y * fast_tanhf(x4.y + q4.y);
    e += s4.z * fast_tanhf(x4.z + q4.z);
    e += s4.w * fast_tanhf(x4.w + q4.w);
  }

  // Block softmax over the 256 e-values (axis i).
  red[t] = e;
  __syncthreads();
  for (int s = 128; s > 0; s >>= 1) {
    if (t < s) red[t] = fmaxf(red[t], red[t + s]);
    __syncthreads();
  }
  const float mx = red[0];
  __syncthreads();
  const float p = __expf(e - mx);
  red[t] = p;
  __syncthreads();
  for (int s = 128; s > 0; s >>= 1) {
    if (t < s) red[t] += red[t + s];
    __syncthreads();
  }
  out[(size_t)(b * 256 + t) * 256 + j] = p / red[0];
}

// ============================= Launch ======================================
extern "C" void kernel_launch(void* const* d_in, const int* in_sizes, int n_in,
                              void* d_out, int out_size, void* d_ws, size_t ws_size,
                              hipStream_t stream) {
  (void)in_sizes; (void)n_in; (void)out_size; (void)ws_size;
  const float* inputs  = (const float*)d_in[0];  // (256, 8, 512) f32
  const float* attn_W  = (const float*)d_in[1];  // (512, 1024)   f32
  const float* attn_b  = (const float*)d_in[2];  // (512,)        f32
  const float* score_W = (const float*)d_in[3];  // (1, 512)      f32
  const float* score_b = (const float*)d_in[4];  // (1,)          f32
  float* out = (float*)d_out;                    // (8, 256, 256) f32
  float* P   = (float*)d_ws;                     // needs 2*2048*512*4 = 8 MB

  // 8192 tiles / 8 waves per block = 1024 blocks
  proj_wmma<<<1024, 256, 0, stream>>>(inputs, attn_W, P);
  // one block per (b, j) column
  energy_softmax<<<2048, 256, 0, stream>>>(P, attn_b, score_W, score_b, out);
}
